// GumbelVectorQuantizer_28638841930057
// MI455X (gfx1250) — compile-verified, matrix operand order checked by reading, not run on hardware
//
#include <hip/hip_runtime.h>
#include <hip/hip_bf16.h>

// ---------------------------------------------------------------------------
// GumbelVectorQuantizer, fused for MI455X (gfx1250, wave32, WMMA f16 16x16x32)
//   B=32 T=1536 C=768 G=2 Vg=160 (GV=320) TEMP=2
//   out = [quantized (B,T,C) f32][codes (B,T,G) i32][probs (B,T,G,Vg) f32]
// ---------------------------------------------------------------------------

typedef __attribute__((ext_vector_type(16))) _Float16 v16h;
typedef __attribute__((ext_vector_type(8)))  _Float16 v8h;
typedef __attribute__((ext_vector_type(8)))  float    v8f;

#define M_TOTAL   49152      // B*T
#define C_DIM     768
#define GV_DIM    320        // G*Vg
#define VG_DIM    160
#define G_DIM     2

#define XLDS_STRIDE 776      // halfs per row (768 + 8 pad), 1552B = 97*16 -> 16B aligned rows
#define LLDS_STRIDE 328      // fp32 logits stride; also fp16 probs stride (656B = 41*16)

#define QUANT_ELEMS  37748736u   // 32*1536*768
#define CODES_ELEMS  98304u      // 32*1536*2
#define CODES_OFF    37748736u
#define PROBS_OFF    37847040u   // QUANT + CODES

// Load a 16x32 f16 WMMA A/B fragment when the source is K-contiguous per
// row/column ("rowStart" points at element K = k0 of this lane's row/col).
// CDNA5 16-bit fragment layout: lanes 0-15 hold K = {0..7, 16..23},
// lanes 16-31 hold K = {8..15, 24..31}; koff (0 or 8) already folded in by
// the caller so we just read two aligned 16-byte runs.
__device__ __forceinline__ v16h load_frag_kcontig(const _Float16* p) {
  v8h lo = *(const v8h*)(p);
  v8h hi = *(const v8h*)(p + 16);
  v16h r;
#pragma unroll
  for (int i = 0; i < 8; ++i) { r[i] = lo[i]; r[8 + i] = hi[i]; }
  return r;
}

// ---------------------------------------------------------------------------
// Prep kernel: fp32 -> fp16 with transpose so GEMM B-operands are K-contiguous.
//   Wt [n, k]  = W_proj[k, n]        (320 x 768 halfs)
//   CBt[c, gv] = codebook[gv, c]     (768 x 320 halfs)
// ---------------------------------------------------------------------------
__global__ void gvq_cvt_kernel(const float* __restrict__ W,
                               const float* __restrict__ CB,
                               _Float16* __restrict__ Wt,
                               _Float16* __restrict__ CBt) {
  int i = blockIdx.x * 256 + threadIdx.x;       // 0 .. 245759
  if (i < C_DIM * GV_DIM) {
    int k  = i / GV_DIM, n = i % GV_DIM;        // W is (C, GV) row-major
    Wt[(size_t)n * C_DIM + k] = (_Float16)W[i];
    int gv = i / C_DIM, c = i % C_DIM;          // CB is (GV, C) row-major
    CBt[(size_t)c * GV_DIM + gv] = (_Float16)CB[i];
  }
}

// ---------------------------------------------------------------------------
// Main fused kernel: one workgroup (320 threads = 10 wave32) per 16 rows.
// ---------------------------------------------------------------------------
__global__ void __launch_bounds__(320)
gvq_main_kernel(const float* __restrict__ x,
                const float* __restrict__ gumbel,
                const float* __restrict__ bias,
                const _Float16* __restrict__ Wt,
                const _Float16* __restrict__ CBt,
                float* __restrict__ quant,
                int*   __restrict__ codes,
                float* __restrict__ probs_out) {
  __shared__ __align__(16) _Float16 x_lds   [16 * XLDS_STRIDE]; // x tile, fp16
  __shared__ __align__(16) float    logit_f [16 * LLDS_STRIDE]; // logits, fp32
  __shared__ __align__(16) _Float16 prob_h  [16 * LLDS_STRIDE]; // probs, fp16

  const int tid  = threadIdx.x;
  const int wave = tid >> 5;
  const int lane = tid & 31;
  const int m0   = blockIdx.x * 16;

  // ---- Stage x tile (16 x 768) fp32 -> fp16 LDS --------------------------
  for (int t = tid; t < 16 * C_DIM; t += 320) {
    int r = t / C_DIM, c = t % C_DIM;
    x_lds[r * XLDS_STRIDE + c] = (_Float16)x[(size_t)(m0 + r) * C_DIM + c];
  }
  __syncthreads();

  const int fragRow = lane & 15;            // M (or N) index within tile
  const int koff    = (lane >> 4) * 8;      // per-lane K sub-offset

  // ---- Phase 1: logits(16x320) = x(16x768) @ W(768x320), 2 N-tiles/wave --
  {
    const int nt0 = wave * 2;
    v8f acc0 = {}, acc1 = {};
    const _Float16* arow = &x_lds[fragRow * XLDS_STRIDE + koff];
    const _Float16* b0   = Wt + (size_t)(nt0 * 16 + fragRow)       * C_DIM + koff;
    const _Float16* b1   = Wt + (size_t)((nt0 + 1) * 16 + fragRow) * C_DIM + koff;
    for (int k0 = 0; k0 < C_DIM; k0 += 32) {
      v16h a  = load_frag_kcontig(arow + k0);
      v16h w0 = load_frag_kcontig(b0 + k0);
      v16h w1 = load_frag_kcontig(b1 + k0);
      acc0 = __builtin_amdgcn_wmma_f32_16x16x32_f16(false, a, false, w0,
                                                    (short)0, acc0, false, false);
      acc1 = __builtin_amdgcn_wmma_f32_16x16x32_f16(false, a, false, w1,
                                                    (short)0, acc1, false, false);
    }
    float bias0 = bias[nt0 * 16 + fragRow];
    float bias1 = bias[(nt0 + 1) * 16 + fragRow];
    const int rbase = (lane >> 4) * 8;       // C/D layout: rows r / r+8
#pragma unroll
    for (int r = 0; r < 8; ++r) {
      int row = rbase + r;
      logit_f[row * LLDS_STRIDE + nt0 * 16 + fragRow]       = acc0[r] + bias0;
      logit_f[row * LLDS_STRIDE + (nt0 + 1) * 16 + fragRow] = acc1[r] + bias1;
    }
  }
  __syncthreads();

  // ---- Phase 2: gumbel-softmax, argmax, emit probs(f32 HBM, f16 LDS) -----
  // 32 problems (16 rows x 2 groups), each handled by one whole wave.
  for (int p = wave; p < 32; p += 10) {
    const int row = p >> 1, g = p & 1;
    const int baseCol = g * VG_DIM;
    const size_t grow = (size_t)(m0 + row) * GV_DIM + baseCol;
    float v[5];
    float mv; int mi;
#pragma unroll
    for (int i = 0; i < 5; ++i) {
      int idx = lane + i * 32;                       // 5*32 = 160
      float lg = logit_f[row * LLDS_STRIDE + baseCol + idx];
      v[i] = (lg + gumbel[grow + idx]) * 0.5f;       // /TEMP
      if (i == 0) { mv = v[0]; mi = idx; }
      else if (v[i] > mv) { mv = v[i]; mi = idx; }
    }
#pragma unroll
    for (int off = 16; off > 0; off >>= 1) {         // wave32 arg-max
      float ov = __shfl_xor(mv, off, 32);
      int   oi = __shfl_xor(mi, off, 32);
      if (ov > mv || (ov == mv && oi < mi)) { mv = ov; mi = oi; }
    }
    float e[5], s = 0.f;
#pragma unroll
    for (int i = 0; i < 5; ++i) { e[i] = __expf(v[i] - mv); s += e[i]; }
#pragma unroll
    for (int off = 16; off > 0; off >>= 1) s += __shfl_xor(s, off, 32);
    float inv = 1.0f / s;
#pragma unroll
    for (int i = 0; i < 5; ++i) {
      int idx = lane + i * 32;
      float pr = e[i] * inv;
      probs_out[grow + idx] = pr;
      prob_h[row * LLDS_STRIDE + baseCol + idx] = (_Float16)pr;
    }
    if (lane == 0) codes[(size_t)(m0 + row) * G_DIM + g] = mi;
  }
  __syncthreads();

  // ---- Phase 3: quantized(16x768) = probs(16x320) @ CB(320x768) ----------
  for (int nt = wave; nt < 48; nt += 10) {           // 48 N-tiles of 16
    v8f acc = {};
    const _Float16* arow = &prob_h[fragRow * LLDS_STRIDE + koff];
    const _Float16* brow = CBt + (size_t)(nt * 16 + fragRow) * GV_DIM + koff;
    for (int k0 = 0; k0 < GV_DIM; k0 += 32) {        // 10 WMMA steps
      v16h a = load_frag_kcontig(arow + k0);
      v16h b = load_frag_kcontig(brow + k0);
      acc = __builtin_amdgcn_wmma_f32_16x16x32_f16(false, a, false, b,
                                                   (short)0, acc, false, false);
    }
    const int rbase = (lane >> 4) * 8;
#pragma unroll
    for (int r = 0; r < 8; ++r) {
      int row = rbase + r;
      quant[(size_t)(m0 + row) * C_DIM + nt * 16 + fragRow] = acc[r];
    }
  }
}

extern "C" void kernel_launch(void* const* d_in, const int* in_sizes, int n_in,
                              void* d_out, int out_size, void* d_ws, size_t ws_size,
                              hipStream_t stream) {
  const float* x      = (const float*)d_in[0];   // (B,T,C)
  const float* gumbel = (const float*)d_in[1];   // (B,T,G,Vg)
  const float* W      = (const float*)d_in[2];   // (C, G*Vg)
  const float* b      = (const float*)d_in[3];   // (G*Vg,)
  const float* CB     = (const float*)d_in[4];   // (G,Vg,C)

  _Float16* Wt  = (_Float16*)d_ws;                               // 320x768 f16
  _Float16* CBt = (_Float16*)((char*)d_ws + (size_t)C_DIM * GV_DIM * 2); // 768x320 f16

  float* quant = (float*)d_out;
  int*   codes = (int*)d_out + CODES_OFF;
  float* probs = (float*)d_out + PROBS_OFF;

  // 245760 elements -> 960 blocks x 256
  hipLaunchKernelGGL(gvq_cvt_kernel, dim3(960), dim3(256), 0, stream,
                     W, CB, Wt, CBt);
  // 49152/16 = 3072 row tiles, 320 threads (10 wave32) each
  hipLaunchKernelGGL(gvq_main_kernel, dim3(M_TOTAL / 16), dim3(320), 0, stream,
                     x, gumbel, b, Wt, CBt, quant, codes, probs);
}
